// GraphUnet_46600395162313
// MI455X (gfx1250) — matrix operand order repaired and checked by reading, not compiled, not run on hardware
//
#include <hip/hip_runtime.h>
#include <hip/hip_bf16.h>

// ---------------------------------------------------------------------------
// Types for CDNA5 WMMA (gfx1250, wave32)
// ---------------------------------------------------------------------------
typedef __attribute__((ext_vector_type(16))) __bf16 v16bf;
typedef __attribute__((ext_vector_type(8)))  float  v8f;

__device__ inline unsigned short f2bf(float x) {
    union { float f; unsigned int u; } c; c.f = x;
    unsigned int u = c.u;
    unsigned int r = u + 0x7fffu + ((u >> 16) & 1u);   // round-to-nearest-even
    return (unsigned short)(r >> 16);
}
__device__ inline __bf16 bits2bf(unsigned short u) { return __builtin_bit_cast(__bf16, u); }

union Pack8 {
    unsigned short u[8];
    uint4 v;
};

// Load 8 consecutive K-elements as bf16 bits (vectorized, no per-elem guards).
__device__ inline Pack8 load8(const float* __restrict__ p) {
    const float4* q = (const float4*)p;
    float4 a = q[0], b = q[1];
    Pack8 r;
    r.u[0] = f2bf(a.x); r.u[1] = f2bf(a.y); r.u[2] = f2bf(a.z); r.u[3] = f2bf(a.w);
    r.u[4] = f2bf(b.x); r.u[5] = f2bf(b.y); r.u[6] = f2bf(b.z); r.u[7] = f2bf(b.w);
    return r;
}
__device__ inline Pack8 load8(const unsigned short* __restrict__ p) {
    Pack8 r;
    r.v = *(const uint4*)p;     // raw bf16 bits pass through
    return r;
}

// ---------------------------------------------------------------------------
// WMMA GEMM:  C[M,N] = act( A[M,K] @ B[K,N] (+ bias) )
// REQUIRES: K % 32 == 0, lda >= K (callers pad att K-dim), N % 4 == 0,
//           B rows [0,K) readable (pad rows multiplied by A-zeros).
// Tile 64 x BN_ x 32, 256 threads = 8 waves. BN_ covers the whole output
// width (grid.y == 1) so the big A stream (bf16 att) is read exactly once.
// Each wave: 16 x (BN_/2) of C via NT = BN_/32 WMMAs per k-step.
// ---------------------------------------------------------------------------
#define BM 64
#define BK 32

template <typename TA, int BN_>
__global__ __launch_bounds__(256) void gemm_wmma(
    const TA* __restrict__ A, const float* __restrict__ B,
    const float* __restrict__ bias, float* __restrict__ C,
    int M, int N, int K, int lda, int ldb, int ldc, int act)
{
    constexpr int NT = BN_ / 32;              // wmma tiles per wave per k-step
    __shared__ unsigned short sA[BM][BK];     // [m][k], k contiguous
    __shared__ unsigned short sB[BN_][BK];    // transposed: [n][k], k contiguous

    const int tid  = threadIdx.x;
    const int lane = tid & 31;
    const int wave = tid >> 5;
    const int hl   = lane & 15;
    const int hi   = lane >> 4;

    const int rowBase = blockIdx.x * BM;
    const int colBase = blockIdx.y * BN_;

    const int wrow = wave & 3;                // 4 M-groups of 16
    const int wcol = wave >> 2;               // 2 N-halves of BN_/2
    const int mo   = wrow * 16;
    const int no   = wcol * (BN_ / 2);

    // A-tile mapping: one row slice of 8 k's per thread
    const int ar  = tid >> 2;                 // 0..63
    const int ak  = (tid & 3) * 8;            // 0,8,16,24
    const int agr = rowBase + ar;

    // B-tile mapping: one k-row, NT groups of 4 columns per thread
    const int bk  = tid >> 3;                 // 0..31
    const int bn0 = (tid & 7) * 4;            // 0..28

    v8f acc[NT];
    const v8f zero8 = {0.f, 0.f, 0.f, 0.f, 0.f, 0.f, 0.f, 0.f};
#pragma unroll
    for (int t = 0; t < NT; ++t) acc[t] = zero8;

    const int kTiles = K >> 5;                // K % 32 == 0 guaranteed
    for (int kt = 0; kt < kTiles; ++kt) {
        const int k0 = kt * BK;

        // ---- A tile: 1 guard + vector load per thread ----
        {
            Pack8 p;
            if (agr < M) {
                p = load8(A + (size_t)agr * lda + (k0 + ak));
            } else {
                p.v = make_uint4(0u, 0u, 0u, 0u);
            }
            *(uint4*)&sA[ar][ak] = p.v;       // 16B aligned ds_store_b128
        }
        // ---- B tile: NT float4 loads per thread (coalesced on n) ----
#pragma unroll
        for (int i = 0; i < NT; ++i) {
            int n4 = bn0 + 32 * i;
            int gn = colBase + n4;
            float4 v = make_float4(0.f, 0.f, 0.f, 0.f);
            if (gn < N)                       // N % 4 == 0 -> whole vector in/out
                v = *(const float4*)(B + (size_t)(k0 + bk) * ldb + gn);
            sB[n4 + 0][bk] = f2bf(v.x);
            sB[n4 + 1][bk] = f2bf(v.y);
            sB[n4 + 2][bk] = f2bf(v.z);
            sB[n4 + 3][bk] = f2bf(v.w);
        }
        // speculative prefetch of next B tile (-> global_prefetch_b8)
        if (k0 + BK < K) {
            int pn = colBase + bn0;
            if (pn < N)
                __builtin_prefetch(&B[(size_t)(k0 + BK + bk) * ldb + pn], 0, 1);
        }
        __syncthreads();

        // ---- build A fragment: 16x32 bf16, ISA 05_wmma layout ----
        v16bf af;
        {
            int r = mo + hl;
#pragma unroll
            for (int j = 0; j < 4; ++j) {          // VGPR 0..3: K = hi*8 + 2j
                int k = hi * 8 + 2 * j;
                af[2 * j]     = bits2bf(sA[r][k]);
                af[2 * j + 1] = bits2bf(sA[r][k + 1]);
            }
#pragma unroll
            for (int j = 0; j < 4; ++j) {          // VGPR 4..7: K = 16 + hi*8 + 2j
                int k = 16 + hi * 8 + 2 * j;
                af[8 + 2 * j]     = bits2bf(sA[r][k]);
                af[8 + 2 * j + 1] = bits2bf(sA[r][k + 1]);
            }
        }
        // ---- NT WMMAs, one A fragment reused across the whole N-half ----
#pragma unroll
        for (int t = 0; t < NT; ++t) {
            int nn = no + t * 16 + hl;             // column this lane owns
            int kb = hi * 8;                       // K = hi*8 + 2j, then 16 + hi*8 + 2j (matches A)
            v16bf bf;
#pragma unroll
            for (int j = 0; j < 8; ++j) {
                int k = (j < 4) ? (kb + 2 * j) : (16 + kb + 2 * (j - 4));
                bf[2 * j]     = bits2bf(sB[nn][k]);
                bf[2 * j + 1] = bits2bf(sB[nn][k + 1]);
            }
            acc[t] = __builtin_amdgcn_wmma_f32_16x16x32_bf16(
                false, af, false, bf, (short)0, acc[t], false, false);
        }
        __syncthreads();
    }

    // ---- epilogue: bias + leaky-relu + store (C/D lane layout) ----
#pragma unroll
    for (int t = 0; t < NT; ++t) {
        int gc = colBase + no + t * 16 + hl;
#pragma unroll
        for (int v = 0; v < 8; ++v) {
            int gr = rowBase + mo + v + 8 * hi;
            if (gr < M && gc < N) {
                float val = acc[t][v];
                if (bias) val += bias[gc];
                if (act)  val = val > 0.f ? val : 0.01f * val;
                C[(size_t)gr * ldc + gc] = val;
            }
        }
    }
}

// ---------------------------------------------------------------------------
// Bit-pack an fp32 0/1 adjacency into a 32x smaller mask via wave32 ballot.
// grid: ((n+255)/256, n)
// ---------------------------------------------------------------------------
__global__ __launch_bounds__(256) void pack_mask(
    const float* __restrict__ adj, unsigned int* __restrict__ pm, int n, int nw)
{
    int i = blockIdx.y;
    int j = blockIdx.x * 256 + threadIdx.x;
    bool on = (j < n) && (adj[(size_t)i * n + j] > 0.f);
    unsigned long long bal = __ballot(on);          // wave32: low 32 bits
    if ((threadIdx.x & 31) == 0 && j < n)
        pm[(size_t)i * nw + (j >> 5)] = (unsigned int)bal;
}

// ---------------------------------------------------------------------------
// Pooled packed mask directly from the parent packed mask (no fp32 adjacency
// is ever materialized for pooled graphs):  out[r,c] = parent[idx[r], idx[c]]
// grid: ((k+255)/256, k)
// ---------------------------------------------------------------------------
__global__ __launch_bounds__(256) void gather_mask(
    const unsigned int* __restrict__ pm, int pnw, const int* __restrict__ idx,
    unsigned int* __restrict__ om, int k, int onw)
{
    int r = blockIdx.y;
    int c = blockIdx.x * 256 + threadIdx.x;
    bool on = false;
    if (c < k) {
        int pr = idx[r], pc = idx[c];
        on = (pm[(size_t)pr * pnw + (pc >> 5)] >> (pc & 31)) & 1u;
    }
    unsigned long long bal = __ballot(on);
    if ((threadIdx.x & 31) == 0 && c < k)
        om[(size_t)r * onw + (c >> 5)] = (unsigned int)bal;
}

// ---------------------------------------------------------------------------
// f[i] = h[i,:]·phiF ; s[i] = h[i,:]·phiS   (one wave32 per row)
// ---------------------------------------------------------------------------
__global__ __launch_bounds__(256) void gemv_fs(
    const float* __restrict__ h, const float* __restrict__ phiF,
    const float* __restrict__ phiS, float* __restrict__ f,
    float* __restrict__ s, int n, int d)
{
    int wave = threadIdx.x >> 5, lane = threadIdx.x & 31;
    int row = blockIdx.x * 8 + wave;
    if (row >= n) return;
    float a = 0.f, b = 0.f;
    for (int k = lane; k < d; k += 32) {
        float hv = h[(size_t)row * d + k];
        a += hv * phiF[k];
        b += hv * phiS[k];
    }
#pragma unroll
    for (int off = 16; off > 0; off >>= 1) {
        a += __shfl_down(a, off, 32);
        b += __shfl_down(b, off, 32);
    }
    if (lane == 0) { f[row] = a; s[row] = b; }
}

// ---------------------------------------------------------------------------
// Row-wise masked softmax -> bf16 attention row (leading dim ldatt, pad
// columns [n, ldatt) zero-filled so the WMMA GEMM needs no ragged-K guards).
// Adjacency comes as a bit-packed mask row (<= 768 B) staged in LDS; the
// only n-sized HBM stream left here is the bf16 att write.
// ---------------------------------------------------------------------------
__global__ __launch_bounds__(256) void row_softmax(
    const unsigned int* __restrict__ msk, int nw,
    const float* __restrict__ f, const float* __restrict__ s,
    unsigned short* __restrict__ att, int n, int ldatt)
{
    __shared__ unsigned int mw[192];          // up to 6144 columns
    __shared__ float red[256];
    const int i = blockIdx.x, tid = threadIdx.x;
    const float fi = f[i];

    for (int w = tid; w < nw; w += 256) mw[w] = msk[(size_t)i * nw + w];
    __syncthreads();

    float m = -3.402823e38f;
    for (int j = tid; j < n; j += 256) {
        bool on = ((mw[j >> 5] >> (j & 31)) & 1u) || (j == i);
        if (on) {
            float v = fi + s[j];
            v = v > 0.f ? v : 0.01f * v;
            m = fmaxf(m, v);
        }
    }
    red[tid] = m; __syncthreads();
    for (int o = 128; o > 0; o >>= 1) {
        if (tid < o) red[tid] = fmaxf(red[tid], red[tid + o]);
        __syncthreads();
    }
    m = red[0]; __syncthreads();

    float sum = 0.f;
    for (int j = tid; j < n; j += 256) {
        bool on = ((mw[j >> 5] >> (j & 31)) & 1u) || (j == i);
        if (on) {
            float v = fi + s[j];
            v = v > 0.f ? v : 0.01f * v;
            sum += __expf(v - m);
        }
    }
    red[tid] = sum; __syncthreads();
    for (int o = 128; o > 0; o >>= 1) {
        if (tid < o) red[tid] += red[tid + o];
        __syncthreads();
    }
    const float inv = 1.f / red[0];

    for (int j = tid; j < ldatt; j += 256) {
        float e = 0.f;
        if (j < n) {
            bool on = ((mw[j >> 5] >> (j & 31)) & 1u) || (j == i);
            if (on) {
                float v = fi + s[j];
                v = v > 0.f ? v : 0.01f * v;
                e = __expf(v - m) * inv;
            }
        }
        att[(size_t)i * ldatt + j] = f2bf(e);
    }
}

// ---------------------------------------------------------------------------
// Pool scores: sigmoid((x@Wp + bp)/100)
// ---------------------------------------------------------------------------
__global__ __launch_bounds__(256) void pool_scores(
    const float* __restrict__ x, const float* __restrict__ Wp,
    const float* __restrict__ bp, float* __restrict__ sc, int n, int d)
{
    int wave = threadIdx.x >> 5, lane = threadIdx.x & 31;
    int row = blockIdx.x * 8 + wave;
    if (row >= n) return;
    float a = 0.f;
    for (int k = lane; k < d; k += 32) a += x[(size_t)row * d + k] * Wp[k];
#pragma unroll
    for (int off = 16; off > 0; off >>= 1) a += __shfl_down(a, off, 32);
    if (lane == 0) {
        float v = (a + bp[0]) * 0.01f;
        sc[row] = 1.f / (1.f + __expf(-v));
    }
}

// ---------------------------------------------------------------------------
// Top-k via single-block bitonic sort of 8192 packed (key,idx) pairs in LDS.
// Descending score, ties broken by ascending index (stable like lax.top_k).
// ---------------------------------------------------------------------------
__global__ __launch_bounds__(1024) void topk_sort(
    const float* __restrict__ sc, int n, int k,
    int* __restrict__ idx, float* __restrict__ vals)
{
    const int MP = 8192, T = 1024;
    __shared__ unsigned long long keys[8192];
    const int tid = threadIdx.x;
    for (int e = tid; e < MP; e += T) {
        unsigned int key2 = 0xFFFFFFFFu;           // padding -> sorts last
        if (e < n) {
            union { float f; unsigned int u; } c; c.f = sc[e];
            unsigned int u = c.u;
            u = (u & 0x80000000u) ? ~u : (u | 0x80000000u);  // order-preserving
            key2 = ~u;                                       // invert: desc score
        }
        keys[e] = ((unsigned long long)key2 << 32) | (unsigned int)e;
    }
    __syncthreads();
    for (int size = 2; size <= MP; size <<= 1) {
        for (int stride = size >> 1; stride > 0; stride >>= 1) {
            for (int e = tid; e < MP; e += T) {
                int j = e ^ stride;
                if (j > e) {
                    bool up = ((e & size) == 0);
                    unsigned long long a = keys[e], b = keys[j];
                    if ((a > b) == up) { keys[e] = b; keys[j] = a; }
                }
            }
            __syncthreads();
        }
    }
    for (int e = tid; e < k; e += T) {
        int id = (int)(keys[e] & 0xFFFFFFFFu);
        idx[e]  = id;
        vals[e] = sc[id];
    }
}

// ---------------------------------------------------------------------------
// Data-movement helpers
// ---------------------------------------------------------------------------
__global__ void gather_rows(const float* __restrict__ x, const int* __restrict__ idx,
                            const float* __restrict__ vals, float* __restrict__ out,
                            int d)
{
    int j = blockIdx.x;
    float v = vals[j];
    const float* src = x + (size_t)idx[j] * d;
    float* dst = out + (size_t)j * d;
    for (int c = threadIdx.x; c < d; c += blockDim.x) dst[c] = src[c] * v;
}

__global__ void scatter_rows(const float* __restrict__ x, const int* __restrict__ idx,
                             float* __restrict__ out, int d)
{
    int j = blockIdx.x;
    const float* src = x + (size_t)j * d;
    float* dst = out + (size_t)idx[j] * d;
    for (int c = threadIdx.x; c < d; c += blockDim.x) dst[c] = src[c];
}

__global__ void fill_zero(float* __restrict__ p, size_t count)
{
    size_t i = (size_t)blockIdx.x * blockDim.x + threadIdx.x;
    if (i < count) p[i] = 0.f;
}

__global__ void add_inplace(float* __restrict__ a, const float* __restrict__ b, size_t count)
{
    size_t i = (size_t)blockIdx.x * blockDim.x + threadIdx.x;
    if (i < count) a[i] += b[i];
}

__global__ void concat_rows(const float* __restrict__ a, const float* __restrict__ b,
                            float* __restrict__ out, int d)
{
    int i = blockIdx.x;
    float* dst = out + (size_t)i * 2 * d;
    const float* sa = a + (size_t)i * d;
    const float* sb = b + (size_t)i * d;
    for (int c = threadIdx.x; c < d; c += blockDim.x) {
        dst[c]     = sa[c];
        dst[d + c] = sb[c];
    }
}

// ---------------------------------------------------------------------------
// One GAT block: h = x@W+b ; f,s = h@phi ; att = softmax(mask(leaky(f+s)))
// out = leaky(att @ h)   (att kept in bf16 with K padded to 32)
// BN_ templated so d=320 layers run with grid.y==1 (att read exactly once).
// ---------------------------------------------------------------------------
template <int BN_>
static void gat_t(hipStream_t st, const unsigned int* msk, int nw, int n,
                  const float* x, int inDim,
                  const float* W, const float* b, const float* phi, int d,
                  float* h, unsigned short* att, float* f, float* s, float* out)
{
    const int npad = (n + 31) & ~31;
    dim3 g1((n + BM - 1) / BM, (d + BN_ - 1) / BN_);
    gemm_wmma<float, BN_><<<g1, 256, 0, st>>>(x, W, b, h, n, d, inDim, inDim, d, d, 0);
    gemv_fs<<<(n + 7) / 8, 256, 0, st>>>(h, phi, phi + d, f, s, n, d);
    row_softmax<<<n, 256, 0, st>>>(msk, nw, f, s, att, n, npad);
    gemm_wmma<unsigned short, BN_><<<g1, 256, 0, st>>>(att, h, nullptr, out,
                                                       n, d, npad, npad, d, d, 1);
}

static void gat(hipStream_t st, const unsigned int* msk, int nw, int n,
                const float* x, int inDim,
                const float* W, const float* b, const float* phi, int d,
                float* h, unsigned short* att, float* f, float* s, float* out)
{
    if (d == 320)
        gat_t<320>(st, msk, nw, n, x, inDim, W, b, phi, d, h, att, f, s, out);
    else
        gat_t<128>(st, msk, nw, n, x, inDim, W, b, phi, d, h, att, f, s, out);
}

// ---------------------------------------------------------------------------
// Host orchestration of the full Graph U-Net
// ---------------------------------------------------------------------------
extern "C" void kernel_launch(void* const* d_in, const int* in_sizes, int n_in,
                              void* d_out, int out_size, void* d_ws, size_t ws_size,
                              hipStream_t stream)
{
    (void)in_sizes; (void)n_in; (void)out_size; (void)ws_size;

    const int n0 = 6144, D = 320, IND = 128, OUTD = 128;
    const int n1 = 4915;              // int(0.8 * 6144)
    const int n2 = 2949;              // int(0.6 * 4915)
    const int nw0 = (n0 + 31) >> 5;   // packed-mask words per row
    const int nw1 = (n1 + 31) >> 5;
    const int nw2 = (n2 + 31) >> 5;

    const float* adj0      = (const float*)d_in[0];
    const float* x0        = (const float*)d_in[1];
    const float* start_W   = (const float*)d_in[2];
    const float* start_b   = (const float*)d_in[3];
    const float* start_phi = (const float*)d_in[4];
    const float* pg0_W     = (const float*)d_in[5];
    const float* pg0_b     = (const float*)d_in[6];
    const float* pg0_phi   = (const float*)d_in[7];
    const float* pg1_W     = (const float*)d_in[8];
    const float* pg1_b     = (const float*)d_in[9];
    const float* pg1_phi   = (const float*)d_in[10];
    const float* pool0_W   = (const float*)d_in[11];
    const float* pool0_b   = (const float*)d_in[12];
    const float* pool1_W   = (const float*)d_in[13];
    const float* pool1_b   = (const float*)d_in[14];
    const float* lat_W     = (const float*)d_in[15];
    const float* lat_b     = (const float*)d_in[16];
    const float* lat_phi   = (const float*)d_in[17];
    const float* ug0_W     = (const float*)d_in[18];
    const float* ug0_b     = (const float*)d_in[19];
    const float* ug0_phi   = (const float*)d_in[20];
    const float* ug1_W     = (const float*)d_in[21];
    const float* ug1_b     = (const float*)d_in[22];
    const float* ug1_phi   = (const float*)d_in[23];
    const float* end_W     = (const float*)d_in[24];
    const float* end_b     = (const float*)d_in[25];
    const float* end_phi   = (const float*)d_in[26];

    float* OUT   = (float*)d_out;                   // (n0, 128)
    float* START = OUT + (size_t)n0 * OUTD;         // (n0, 320) start_gcn_outs

    char* ws = (char*)d_ws;
    size_t off = 0;
    auto carve = [&](size_t bytes) -> char* {
        char* p = ws + off;
        off = (off + bytes + 255) & ~(size_t)255;
        return p;
    };
    unsigned short* ATT = (unsigned short*)carve((size_t)n0 * n0 * 2);
    unsigned int* PM0 = (unsigned int*)carve((size_t)n0 * nw0 * 4);
    unsigned int* PM1 = (unsigned int*)carve((size_t)n1 * nw1 * 4);
    unsigned int* PM2 = (unsigned int*)carve((size_t)n2 * nw2 * 4);
    float* H    = (float*)carve((size_t)n0 * D * 4);
    float* XA   = (float*)carve((size_t)n0 * D * 4);
    float* XB   = (float*)carve((size_t)n0 * D * 4);
    float* D0   = (float*)carve((size_t)n0 * D * 4);
    float* D1   = (float*)carve((size_t)n1 * D * 4);
    float* CAT  = (float*)carve((size_t)n0 * 2 * D * 4);
    float* FV   = (float*)carve((size_t)n0 * 4);
    float* SV   = (float*)carve((size_t)n0 * 4);
    float* SC   = (float*)carve((size_t)n0 * 4);
    int*   IDXA = (int*)carve((size_t)n0 * 4);
    float* VALA = (float*)carve((size_t)n0 * 4);
    int*   IDXB = (int*)carve((size_t)n0 * 4);
    float* VALB = (float*)carve((size_t)n0 * 4);

    // One pass over the fp32 adjacency -> 32x smaller packed mask.
    pack_mask<<<dim3((n0 + 255) / 256, n0), 256, 0, stream>>>(adj0, PM0, n0, nw0);

    // start GAT -> START (also the second output segment / original_input)
    gat(stream, PM0, nw0, n0, x0, IND, start_W, start_b, start_phi, D, H, ATT, FV, SV, START);

    // down path 0
    gat(stream, PM0, nw0, n0, START, D, pg0_W, pg0_b, pg0_phi, D, H, ATT, FV, SV, D0);
    pool_scores<<<(n0 + 7) / 8, 256, 0, stream>>>(D0, pool0_W, pool0_b, SC, n0, D);
    topk_sort<<<1, 1024, 0, stream>>>(SC, n0, n1, IDXA, VALA);
    gather_mask<<<dim3((n1 + 255) / 256, n1), 256, 0, stream>>>(PM0, nw0, IDXA, PM1, n1, nw1);
    gather_rows<<<n1, 256, 0, stream>>>(D0, IDXA, VALA, XA, D);

    // down path 1
    gat(stream, PM1, nw1, n1, XA, D, pg1_W, pg1_b, pg1_phi, D, H, ATT, FV, SV, D1);
    pool_scores<<<(n1 + 7) / 8, 256, 0, stream>>>(D1, pool1_W, pool1_b, SC, n1, D);
    topk_sort<<<1, 1024, 0, stream>>>(SC, n1, n2, IDXB, VALB);
    gather_mask<<<dim3((n2 + 255) / 256, n2), 256, 0, stream>>>(PM1, nw1, IDXB, PM2, n2, nw2);
    gather_rows<<<n2, 256, 0, stream>>>(D1, IDXB, VALB, XA, D);

    // bottleneck
    gat(stream, PM2, nw2, n2, XA, D, lat_W, lat_b, lat_phi, D, H, ATT, FV, SV, XB);

    // up path 0 (u = 1)
    {
        size_t cnt = (size_t)n1 * D;
        fill_zero<<<(unsigned)((cnt + 255) / 256), 256, 0, stream>>>(XA, cnt);
        scatter_rows<<<n2, 256, 0, stream>>>(XB, IDXB, XA, D);
        gat(stream, PM1, nw1, n1, XA, D, ug0_W, ug0_b, ug0_phi, D, H, ATT, FV, SV, XB);
        add_inplace<<<(unsigned)((cnt + 255) / 256), 256, 0, stream>>>(XB, D1, cnt);
    }
    // up path 1 (u = 0)
    {
        size_t cnt = (size_t)n0 * D;
        fill_zero<<<(unsigned)((cnt + 255) / 256), 256, 0, stream>>>(XA, cnt);
        scatter_rows<<<n1, 256, 0, stream>>>(XB, IDXA, XA, D);
        gat(stream, PM0, nw0, n0, XA, D, ug1_W, ug1_b, ug1_phi, D, H, ATT, FV, SV, XB);
        add_inplace<<<(unsigned)((cnt + 255) / 256), 256, 0, stream>>>(XB, D0, cnt);
    }

    // end: concat + final GAT (d = 128, inDim = 640)
    concat_rows<<<n0, 256, 0, stream>>>(XB, START, CAT, D);
    gat(stream, PM0, nw0, n0, CAT, 2 * D, end_W, end_b, end_phi, OUTD, H, ATT, FV, SV, OUT);
}
